// GCN_FineTuned_47425028882724
// MI455X (gfx1250) — matrix operand order repaired and checked by reading, not compiled, not run on hardware
//
#include <hip/hip_runtime.h>

typedef __attribute__((ext_vector_type(16))) _Float16 v16h;
typedef __attribute__((ext_vector_type(8)))  _Float16 v8h;
typedef __attribute__((ext_vector_type(8)))  float    v8f;

// ---------------------------------------------------------------------------
// Utility
// ---------------------------------------------------------------------------
__device__ __forceinline__ void atomAddF(float* p, float v) {
  __hip_atomic_fetch_add(p, v, __ATOMIC_RELAXED, __HIP_MEMORY_SCOPE_AGENT);
}

__global__ void fill_f32_kernel(float* __restrict__ p, float val, long n) {
  long i = (long)blockIdx.x * blockDim.x + threadIdx.x;
  if (i < n) p[i] = val;
}

// deg[dst] += 1  (deg pre-filled with 1.0 for the self loop)
__global__ void degree_kernel(const int* __restrict__ dst, float* __restrict__ deg, int E) {
  long e = (long)blockIdx.x * blockDim.x + threadIdx.x;
  if (e < E) atomAddF(deg + dst[e], 1.0f);
}

// dis = rsqrt(deg) in place
__global__ void rsqrt_kernel(float* __restrict__ d, int N) {
  long i = (long)blockIdx.x * blockDim.x + threadIdx.x;
  if (i < N) d[i] = rsqrtf(d[i]);
}

// enorm[e] = dis[src[e]] * dis[dst[e]]
__global__ void enorm_kernel(const int* __restrict__ src, const int* __restrict__ dst,
                             const float* __restrict__ dis, float* __restrict__ en, int E) {
  long e = (long)blockIdx.x * blockDim.x + threadIdx.x;
  if (e < E) en[e] = dis[src[e]] * dis[dst[e]];
}

// ---------------------------------------------------------------------------
// WMMA fragments (wave32, v_wmma_f32_16x16x32_f16, ISA §7.12.2)
// A (16x32 f16): lane L<16 -> M=L, halves = row[k0+0..7] ++ row[k0+16..23];
//                lane L>=16 -> same M, K base +8.  Two contiguous 16B runs.
// B (32x16 f16): lane L<16 -> N=L, K=0..7,16..23; lane L>=16 -> N=L-16, K=8..15,24..31.
//                Staged pre-swizzled: per (kt,nt) fragment, lane-contiguous
//                32B chunks -> loads as 2x ds_load_b128.
// C/D (16x16 f32): VGPR r: lanes 0-15 -> M=r, lanes 16-31 -> M=8+r; N=lane&15.
// ---------------------------------------------------------------------------
__device__ __forceinline__ v16h load_frag_A(const _Float16* __restrict__ sX,
                                            int lane, int m0, int k0, int lda) {
  const int m   = m0 + (lane & 15);
  const int khi = (lane >> 4) << 3;                 // 0 or 8
  const _Float16* row = sX + m * lda + k0 + khi;
  const v8h lo = *(const v8h*)(row);                // K = khi+0..7
  const v8h hi = *(const v8h*)(row + 16);           // K = khi+16..23
  return __builtin_shufflevector(lo, hi, 0, 1, 2, 3, 4, 5, 6, 7,
                                 8, 9, 10, 11, 12, 13, 14, 15);
}

// ---------------------------------------------------------------------------
// Tiled WMMA GEMM: Y[N,F_OUT] = X[N,F_IN] @ W[F_IN,F_OUT]
// fp32 in/out, f16 compute, fp32 accumulate. ROWS rows/block, 1 wave/N-tile.
// W is staged into LDS already swizzled into per-lane fragment order.
// ---------------------------------------------------------------------------
template <int F_IN, int F_OUT, int ROWS>
__global__ void gemm_wmma_kernel(const float* __restrict__ X,
                                 const float* __restrict__ W,
                                 float* __restrict__ Y, int N) {
  constexpr int LDA = F_IN + 8;                     // pad to spread LDS banks
  constexpr int KT  = F_IN / 32;                    // k-tiles
  constexpr int NT  = F_OUT / 16;                   // n-tiles (= waves)
  __shared__ __align__(32) _Float16 sX[ROWS * LDA];
  __shared__ __align__(32) _Float16 sW[F_IN * F_OUT];  // fragment-swizzled

  const int  tid      = threadIdx.x;
  const int  nthreads = blockDim.x;
  const long rowBase  = (long)blockIdx.x * ROWS;

  // Stage X tile (fp32 -> f16), zero-pad OOB rows.
  for (int i = tid; i < ROWS * F_IN; i += nthreads) {
    const int  r  = i / F_IN, c = i % F_IN;
    const long gr = rowBase + r;
    sX[r * LDA + c] = (gr < N) ? (_Float16)X[gr * F_IN + c] : (_Float16)0.0f;
  }
  // Stage W swizzled: sW[((kt*NT+nt)*32 + lane)*16 + h] = W[k, n] with
  //   k = kt*32 + ((lane>>4)<<3) + (h&7) + ((h>>3)<<4),  n = nt*16 + (lane&15).
  for (int i = tid; i < F_IN * F_OUT; i += nthreads) {
    const int h    = i & 15;
    const int ln   = (i >> 4) & 31;
    const int frag = i >> 9;
    const int nt   = frag % NT;
    const int kt   = frag / NT;
    const int k = kt * 32 + ((ln >> 4) << 3) + (h & 7) + ((h >> 3) << 4);
    const int n = nt * 16 + (ln & 15);
    sW[i] = (_Float16)W[k * F_OUT + n];
  }
  __syncthreads();

  const int lane = tid & 31;
  const int wave = tid >> 5;                        // == nt
  const int n0   = wave * 16;
  const _Float16* sWfrag = sW + (size_t)wave * 512 + (size_t)lane * 16;
  const bool full = (rowBase + ROWS) <= (long)N;
  float* Yb = Y + rowBase * F_OUT;                  // 64-bit math once

  const int col = lane & 15;
  const int ro  = (lane >> 4) << 3;                 // 0 or 8

#pragma unroll
  for (int mt = 0; mt < ROWS / 16; ++mt) {
    v8f acc = {};
#pragma unroll
    for (int kt = 0; kt < KT; ++kt) {
      v16h a = load_frag_A(sX, lane, mt * 16, kt * 32, LDA);
      v16h b = *(const v16h*)(sWfrag + (size_t)kt * (NT * 512));
      acc = __builtin_amdgcn_wmma_f32_16x16x32_f16(false, a, false, b,
                                                   (short)0, acc, false, false);
    }
    if (full) {                                     // block-uniform fast path
      float* p = Yb + (mt * 16 + ro) * F_OUT + n0 + col;
#pragma unroll
      for (int r = 0; r < 8; ++r) p[r * F_OUT] = acc[r];
    } else {
#pragma unroll
      for (int r = 0; r < 8; ++r) {
        const long grow = rowBase + mt * 16 + ro + r;
        if (grow < N) Y[grow * F_OUT + n0 + col] = acc[r];
      }
    }
  }
}

// ---------------------------------------------------------------------------
// Edge scatter: agg[dst,:] += h[src,:] * enorm[e]
// F/4 threads per edge; float4 gather of 512B/256B-contiguous rows,
// native f32 atomics into (L2-resident) agg.
// ---------------------------------------------------------------------------
template <int F>
__global__ void edge_agg_kernel(const int* __restrict__ src,
                                const int* __restrict__ dst,
                                const float* __restrict__ enorm,
                                const float* __restrict__ h,
                                float* __restrict__ agg, int E) {
  constexpr int TPE = F / 4;
  const long tid = (long)blockIdx.x * blockDim.x + threadIdx.x;
  const long e   = tid / TPE;
  if (e >= E) return;
  const int f4 = (int)(tid % TPE) * 4;
  const int s = src[e], d = dst[e];
  const float en = enorm[e];
  const float4 hv = *(const float4*)(h + (long)s * F + f4);
  float* ap = agg + (long)d * F + f4;
  atomAddF(ap + 0, hv.x * en);
  atomAddF(ap + 1, hv.y * en);
  atomAddF(ap + 2, hv.z * en);
  atomAddF(ap + 3, hv.w * en);
}

// out = relu( BN( agg + h*snorm + b ) ), in place over agg
template <int F>
__global__ void post_bn_relu_kernel(float* __restrict__ agg,
                                    const float* __restrict__ h,
                                    const float* __restrict__ dis,
                                    const float* __restrict__ b,
                                    const float* __restrict__ g,
                                    const float* __restrict__ be,
                                    const float* __restrict__ m,
                                    const float* __restrict__ v, int N) {
  const long tid = (long)blockIdx.x * blockDim.x + threadIdx.x;
  const long n = tid / F;
  if (n >= N) return;
  const int f = (int)(tid % F);
  float sn = dis[n]; sn *= sn;
  const float pre = agg[n * F + f] + h[n * F + f] * sn + b[f];
  const float y = (pre - m[f]) * rsqrtf(v[f] + 1e-5f) * g[f] + be[f];
  agg[n * F + f] = fmaxf(y, 0.0f);
}

// ---------------------------------------------------------------------------
// Layer 3 (64 -> 2): too skinny for WMMA; VALU is free at 25 MFLOP.
// ---------------------------------------------------------------------------
__global__ void gemm3_kernel(const float* __restrict__ h,
                             const float* __restrict__ W3,
                             float* __restrict__ y, int N) {
  const long n = (long)blockIdx.x * blockDim.x + threadIdx.x;
  if (n >= N) return;
  float a0 = 0.f, a1 = 0.f;
#pragma unroll 8
  for (int k = 0; k < 64; ++k) {
    const float hv = h[n * 64 + k];
    a0 += hv * W3[k * 2 + 0];
    a1 += hv * W3[k * 2 + 1];
  }
  y[n * 2 + 0] = a0;
  y[n * 2 + 1] = a1;
}

__global__ void edge_agg2_kernel(const int* __restrict__ src,
                                 const int* __restrict__ dst,
                                 const float* __restrict__ enorm,
                                 const float* __restrict__ h3,
                                 float* __restrict__ agg, int E) {
  const long e = (long)blockIdx.x * blockDim.x + threadIdx.x;
  if (e >= E) return;
  const int s = src[e], d = dst[e];
  const float en = enorm[e];
  atomAddF(agg + (long)d * 2 + 0, h3[(long)s * 2 + 0] * en);
  atomAddF(agg + (long)d * 2 + 1, h3[(long)s * 2 + 1] * en);
}

__global__ void final_kernel(const float* __restrict__ agg,
                             const float* __restrict__ h3,
                             const float* __restrict__ dis,
                             const float* __restrict__ b3,
                             float* __restrict__ out, int N) {
  const long n = (long)blockIdx.x * blockDim.x + threadIdx.x;
  if (n >= N) return;
  const float sn = dis[n] * dis[n];
  out[n * 2 + 0] = agg[n * 2 + 0] + h3[n * 2 + 0] * sn + b3[0];
  out[n * 2 + 1] = agg[n * 2 + 1] + h3[n * 2 + 1] * sn + b3[1];
}

// ---------------------------------------------------------------------------
// Launch
// ---------------------------------------------------------------------------
static inline int cdiv_i(long a, long b) { return (int)((a + b - 1) / b); }

extern "C" void kernel_launch(void* const* d_in, const int* in_sizes, int n_in,
                              void* d_out, int out_size, void* d_ws, size_t ws_size,
                              hipStream_t stream) {
  const float* x   = (const float*)d_in[0];
  const int*   ei  = (const int*)d_in[1];   // [2, E] int32 (JAX default x64-off)
  const float* W1  = (const float*)d_in[2];
  const float* b1  = (const float*)d_in[3];
  const float* g1  = (const float*)d_in[4];
  const float* be1 = (const float*)d_in[5];
  const float* m1  = (const float*)d_in[6];
  const float* v1  = (const float*)d_in[7];
  const float* W2  = (const float*)d_in[8];
  const float* b2  = (const float*)d_in[9];
  const float* g2  = (const float*)d_in[10];
  const float* be2 = (const float*)d_in[11];
  const float* m2  = (const float*)d_in[12];
  const float* v2  = (const float*)d_in[13];
  const float* W3  = (const float*)d_in[14];
  const float* b3  = (const float*)d_in[15];

  const int N = in_sizes[0] / 64;
  const int E = in_sizes[1] / 2;
  const int* src = ei;
  const int* dst = ei + E;

  float* ws    = (float*)d_ws;
  float* dis   = ws;                         // N
  float* enorm = dis + N;                    // E
  float* bufH  = enorm + E;                  // N*128 (linear outputs)
  float* bufA  = bufH + (size_t)N * 128;     // N*128 (aggregates / activations)
  float* agg3  = bufA + (size_t)N * 128;     // N*2

  const int TB = 256;
  float* out = (float*)d_out;

  // --- degree + normalization ---
  fill_f32_kernel<<<cdiv_i(N, TB), TB, 0, stream>>>(dis, 1.0f, N);
  degree_kernel<<<cdiv_i(E, TB), TB, 0, stream>>>(dst, dis, E);
  rsqrt_kernel<<<cdiv_i(N, TB), TB, 0, stream>>>(dis, N);
  enorm_kernel<<<cdiv_i(E, TB), TB, 0, stream>>>(src, dst, dis, enorm, E);

  // --- layer 1: 64 -> 128 ---
  gemm_wmma_kernel<64, 128, 128>
      <<<cdiv_i(N, 128), 32 * (128 / 16), 0, stream>>>(x, W1, bufH, N);
  fill_f32_kernel<<<cdiv_i((long)N * 128, TB), TB, 0, stream>>>(bufA, 0.0f, (long)N * 128);
  edge_agg_kernel<128>
      <<<cdiv_i((long)E * 32, TB), TB, 0, stream>>>(src, dst, enorm, bufH, bufA, E);
  post_bn_relu_kernel<128>
      <<<cdiv_i((long)N * 128, TB), TB, 0, stream>>>(bufA, bufH, dis, b1, g1, be1, m1, v1, N);

  // --- layer 2: 128 -> 64 ---
  gemm_wmma_kernel<128, 64, 128>
      <<<cdiv_i(N, 128), 32 * (64 / 16), 0, stream>>>(bufA, W2, bufH, N);
  fill_f32_kernel<<<cdiv_i((long)N * 64, TB), TB, 0, stream>>>(bufA, 0.0f, (long)N * 64);
  edge_agg_kernel<64>
      <<<cdiv_i((long)E * 16, TB), TB, 0, stream>>>(src, dst, enorm, bufH, bufA, E);
  post_bn_relu_kernel<64>
      <<<cdiv_i((long)N * 64, TB), TB, 0, stream>>>(bufA, bufH, dis, b2, g2, be2, m2, v2, N);

  // --- layer 3: 64 -> 2 ---
  gemm3_kernel<<<cdiv_i(N, TB), TB, 0, stream>>>(bufA, W3, bufH, N);
  fill_f32_kernel<<<cdiv_i((long)N * 2, TB), TB, 0, stream>>>(agg3, 0.0f, (long)N * 2);
  edge_agg2_kernel<<<cdiv_i(E, TB), TB, 0, stream>>>(src, dst, enorm, bufH, agg3, E);
  final_kernel<<<cdiv_i(N, TB), TB, 0, stream>>>(agg3, bufH, dis, b3, out, N);
}
